// LHL_block_78151224918227
// MI455X (gfx1250) — matrix operand order repaired and checked by reading, not compiled, not run on hardware
//
#include <hip/hip_runtime.h>
#include <cstddef>
#include <cstdint>
#include <math.h>

#define GN0 50000
#define GE0 200000
#define GN1 100000
#define GE1 400000

typedef __attribute__((ext_vector_type(16))) __bf16          v16bf;
typedef __attribute__((ext_vector_type(8)))  float           v8f;
typedef __attribute__((ext_vector_type(4)))  float           f32x4;
typedef __attribute__((ext_vector_type(8)))  unsigned short  u16x8;
typedef __attribute__((ext_vector_type(4)))  unsigned short  u16x4;

__device__ __forceinline__ unsigned short f2bf(float x) {
  unsigned int u = __float_as_uint(x);
  unsigned int r = u + 0x7FFFu + ((u >> 16) & 1u);   // round-to-nearest-even
  return (unsigned short)(r >> 16);
}

// ---------------------------------------------------------------------------
// C[M,N] = A @ W (+bias[N]) (+Dadd[M,N]); A row index = row / a_row_div
// 256 threads = 8 waves; block tile 64x32; wave tile 16x16; K step 32.
// LDS layouts are pre-arranged so each lane's WMMA fragment is two contiguous
// ds_load_b128: A rows stored in linear K order (same as B, k = 16h + s),
// B stored transposed [n][k] (fragment K order is linear).
// ---------------------------------------------------------------------------
__global__ __launch_bounds__(256) void gemm_wmma_kernel(
    const float* __restrict__ A, int lda, int a_row_div,
    const float* __restrict__ W,
    const float* __restrict__ bias,
    const float* __restrict__ Dadd,
    float* __restrict__ C,
    int M, int K, int N)
{
  __shared__ __align__(16) unsigned short As[64][40];  // 80B pitch, 16B aligned
  __shared__ __align__(16) unsigned short Bs[32][40];

  const int bm   = blockIdx.x * 64;
  const int bn   = blockIdx.y * 32;
  const int tid  = threadIdx.x;
  const int lane = tid & 31;
  const int wave = tid >> 5;
  const int wm    = (wave >> 1) << 4;   // 0,16,32,48
  const int wn    = (wave & 1) << 4;    // 0,16
  const int lhalf = lane >> 4;          // 0/1
  const int l16   = lane & 15;

  // --- A staging assignment: thread -> (row, 8-col group), hoisted row math
  const int arow = tid >> 2;                       // 0..63
  const int ag   = tid & 3;                        // col group (8 floats)
  const int apg  = ag;                             // linear K order (matches B fragment order)
  const int agr  = bm + arow;
  const bool a_ok = (agr < M);
  const int agrc  = a_ok ? agr : (M - 1);
  const float* aRow = A + (size_t)(agrc / a_row_div) * lda + ag * 8;

  // --- B staging assignment: thread -> (n col, 4-row K group)
  const int bnc = tid & 31;            // n within tile
  const int br4 = (tid >> 5) << 2;     // k group of 4
  const float* wCol = W + (size_t)br4 * N + bn + bnc;

  v8f acc = {0.f,0.f,0.f,0.f,0.f,0.f,0.f,0.f};

  for (int k0 = 0; k0 < K; k0 += 32) {
    __syncthreads();
    // stage A: two global b128 loads, one LDS b128 store (branchless M guard)
    {
      const float* ap = aRow + k0;
      f32x4 x0 = *(const f32x4*)ap;
      f32x4 x1 = *(const f32x4*)(ap + 4);
      u16x8 h;
      #pragma unroll
      for (int j = 0; j < 4; ++j) h[j]     = f2bf(a_ok ? x0[j] : 0.f);
      #pragma unroll
      for (int j = 0; j < 4; ++j) h[4 + j] = f2bf(a_ok ? x1[j] : 0.f);
      *(u16x8*)&As[arow][apg * 8] = h;
    }
    // stage B transposed: 4 coalesced b32 loads, one LDS b64 store
    {
      const float* wp = wCol + (size_t)k0 * N;
      u16x4 h;
      #pragma unroll
      for (int j = 0; j < 4; ++j) h[j] = f2bf(wp[(size_t)j * N]);
      *(u16x4*)&Bs[bnc][br4] = h;
    }
    __syncthreads();

    // fragments: contiguous 32B per lane (2x ds_load_b128 each)
    union { v16bf v; u16x8 h[2]; } fa, fb;
    fa.h[0] = *(const u16x8*)&As[wm + l16][lhalf * 16];
    fa.h[1] = *(const u16x8*)&As[wm + l16][lhalf * 16 + 8];
    fb.h[0] = *(const u16x8*)&Bs[wn + l16][lhalf * 16];
    fb.h[1] = *(const u16x8*)&Bs[wn + l16][lhalf * 16 + 8];

    acc = __builtin_amdgcn_wmma_f32_16x16x32_bf16(false, fa.v, false, fb.v,
                                                  (short)0, acc, false, false);
  }

  const int gn = bn + wn + l16;
  const float bv = bias ? bias[gn] : 0.f;
  #pragma unroll
  for (int r = 0; r < 8; ++r) {
    int gm = bm + wm + lhalf * 8 + r;   // C layout: VGPR r -> M = r + 8*laneHi
    if (gm < M) {
      float v = acc[r] + bv;
      if (Dadd) v += Dadd[(size_t)gm * N + gn];
      C[(size_t)gm * N + gn] = v;
    }
  }
}

// ---------------------------------------------------------------------------
__global__ void fill_kernel(float* __restrict__ p, size_t n, float v) {
  size_t i = (size_t)blockIdx.x * blockDim.x + threadIdx.x;
  if (i < n) p[i] = v;
}

__device__ __forceinline__ void atomicMaxF(float* addr, float val) {
  if (val >= 0.f) atomicMax((int*)addr, __float_as_int(val));
  else            atomicMin((unsigned int*)addr, __float_as_uint(val));
}

// f = lrelu(FNi[src]+FNj[dst]+FE+bias); logits[e,h] = sum_c f*attn; emax = segmax(dst)
__global__ __launch_bounds__(256) void edge_combine_kernel(
    const float* __restrict__ FNi, const float* __restrict__ FNj,
    const float* __restrict__ FE,  const float* __restrict__ bias,
    const float* __restrict__ attn,
    const int* __restrict__ src, const int* __restrict__ dst,
    float* __restrict__ f_out, float* __restrict__ logits, float* __restrict__ emax,
    int E, int F2, int out_e)
{
  int w    = (int)(((size_t)blockIdx.x * 256 + threadIdx.x) >> 5);
  int lane = threadIdx.x & 31;
  if (w >= E) return;
  int s = src[w], d = dst[w];
  float a0 = 0.f, a1 = 0.f;
  size_t eb = (size_t)w * F2;
  for (int c = lane; c < F2; c += 32) {
    float v = FNi[(size_t)s * F2 + c] + FNj[(size_t)d * F2 + c] + FE[eb + c] + bias[c];
    v = (v > 0.f) ? v : 0.01f * v;           // leaky_relu, slope 0.01
    f_out[eb + c] = v;
    float t = v * attn[c];
    if (c < out_e) a0 += t; else a1 += t;    // H = 2 heads
  }
  #pragma unroll
  for (int off = 16; off > 0; off >>= 1) {
    a0 += __shfl_down(a0, off, 32);
    a1 += __shfl_down(a1, off, 32);
  }
  if (lane == 0) {
    logits[(size_t)w * 2 + 0] = a0;
    logits[(size_t)w * 2 + 1] = a1;
    atomicMaxF(&emax[(size_t)d * 2 + 0], a0);
    atomicMaxF(&emax[(size_t)d * 2 + 1], a1);
  }
}

__global__ void edge_exp_kernel(const float* __restrict__ logits,
                                const float* __restrict__ emax,
                                const int* __restrict__ dst,
                                float* __restrict__ ee, float* __restrict__ denom, int E)
{
  int i = blockIdx.x * blockDim.x + threadIdx.x;
  if (i >= E * 2) return;
  int e = i >> 1, h = i & 1;
  int d = dst[e];
  float v = expf(logits[i] - emax[(size_t)d * 2 + h]);
  ee[i] = v;
  atomicAdd(&denom[(size_t)d * 2 + h], v);
}

// hacc[dst] += Hnode[src] * a[h]
__global__ __launch_bounds__(256) void edge_aggregate_kernel(
    const float* __restrict__ Hn, const float* __restrict__ ee,
    const float* __restrict__ denom,
    const int* __restrict__ src, const int* __restrict__ dst,
    float* __restrict__ hacc, int E, int FN2, int out_n)
{
  int w    = (int)(((size_t)blockIdx.x * 256 + threadIdx.x) >> 5);
  int lane = threadIdx.x & 31;
  if (w >= E) return;
  int s = src[w], d = dst[w];
  float a0 = ee[(size_t)w * 2 + 0] / denom[(size_t)d * 2 + 0];
  float a1 = ee[(size_t)w * 2 + 1] / denom[(size_t)d * 2 + 1];
  for (int c = lane; c < FN2; c += 32) {
    float a = (c < out_n) ? a0 : a1;
    atomicAdd(&hacc[(size_t)d * FN2 + c], Hn[(size_t)s * FN2 + c] * a);
  }
}

// out[r,c] = in[r, c] + in[r, F + c]   (sum over H=2 heads)
__global__ void head_sum_kernel(const float* __restrict__ in,
                                float* __restrict__ out, size_t M, int F)
{
  size_t i = (size_t)blockIdx.x * blockDim.x + threadIdx.x;
  if (i >= M * (size_t)F) return;
  size_t r = i / (size_t)F;
  int c = (int)(i - r * (size_t)F);
  size_t b = r * (size_t)(2 * F);
  out[i] = in[b + c] + in[b + F + c];
}

// ---------------------------------------------------------------------------
// Host orchestration
// ---------------------------------------------------------------------------
struct EgatP { const float *ni,*nj,*fij,*node,*bnode,*attn,*bias; };

// workspace layout (float offsets); HNODE aliases FE, HACC aliases FNI+FNJ
static const size_t oFNI = 0;          // 12.8M
static const size_t oFNJ = 12800000;   // 12.8M
static const size_t oFE  = 25600000;   // 51.2M  (later: HNODE, 25.6M)
static const size_t oF   = 76800000;   // 51.2M
static const size_t oLOG = 128000000;  // 0.8M
static const size_t oEE  = 128800000;  // 0.8M
static const size_t oEMX = 129600000;  // 0.2M
static const size_t oDEN = 129800000;  // 0.2M
static const size_t oNF  = 130000000;  // 6.4M  (stage1 node out)
static const size_t oEFS = 136400000;  // 25.6M (stage1 edge out / lin1 input)
// total 162,000,000 floats = 648 MB

static void run_egat(const float* nfeats, const float* efeats, int e_row_div,
                     const int* src, const int* dst,
                     int Nn, int E, int in_n, int in_e, int out_n, int out_e,
                     const EgatP& p, float* ws,
                     float* node_out, float* edge_out, hipStream_t stream)
{
  const int F2  = 2 * out_e;   // H=2
  const int FN2 = 2 * out_n;
  float* FNI   = ws + oFNI;
  float* FNJ   = ws + oFNJ;
  float* FE    = ws + oFE;
  float* F     = ws + oF;
  float* LOG   = ws + oLOG;
  float* EE    = ws + oEE;
  float* EMX   = ws + oEMX;
  float* DEN   = ws + oDEN;
  float* HNODE = ws + oFE;    // alias: FE dead after combine
  float* HACC  = ws + oFNI;   // alias: FNI+FNJ dead after combine (25.6M contiguous)

  dim3 blk(256);
  // dense GEMMs
  gemm_wmma_kernel<<<dim3((Nn + 63) / 64, F2 / 32), blk, 0, stream>>>(
      nfeats, in_n, 1, p.ni, nullptr, nullptr, FNI, Nn, in_n, F2);
  gemm_wmma_kernel<<<dim3((Nn + 63) / 64, F2 / 32), blk, 0, stream>>>(
      nfeats, in_n, 1, p.nj, nullptr, nullptr, FNJ, Nn, in_n, F2);
  gemm_wmma_kernel<<<dim3((E + 63) / 64, F2 / 32), blk, 0, stream>>>(
      efeats, in_e, e_row_div, p.fij, nullptr, nullptr, FE, E, in_e, F2);

  // accumulator init (must precede combine/exp)
  fill_kernel<<<(Nn * 2 + 255) / 256, blk, 0, stream>>>(EMX, (size_t)Nn * 2, -INFINITY);
  fill_kernel<<<(Nn * 2 + 255) / 256, blk, 0, stream>>>(DEN, (size_t)Nn * 2, 0.f);

  edge_combine_kernel<<<(E + 7) / 8, blk, 0, stream>>>(
      FNI, FNJ, FE, p.bias, p.attn, src, dst, F, LOG, EMX, E, F2, out_e);
  edge_exp_kernel<<<(E * 2 + 255) / 256, blk, 0, stream>>>(LOG, EMX, dst, EE, DEN, E);

  // node projection (overwrites FE slab -> ordered after combine)
  gemm_wmma_kernel<<<dim3((Nn + 63) / 64, FN2 / 32), blk, 0, stream>>>(
      nfeats, in_n, 1, p.node, p.bnode, nullptr, HNODE, Nn, in_n, FN2);

  fill_kernel<<<((size_t)Nn * FN2 + 255) / 256, blk, 0, stream>>>(HACC, (size_t)Nn * FN2, 0.f);
  edge_aggregate_kernel<<<(E + 7) / 8, blk, 0, stream>>>(
      HNODE, EE, DEN, src, dst, HACC, E, FN2, out_n);

  head_sum_kernel<<<((size_t)Nn * out_n + 255) / 256, blk, 0, stream>>>(HACC, node_out, (size_t)Nn, out_n);
  head_sum_kernel<<<((size_t)E * out_e + 255) / 256, blk, 0, stream>>>(F, edge_out, (size_t)E, out_e);
}

static EgatP getP(void* const* d_in, int b) {
  return EgatP{ (const float*)d_in[b+0], (const float*)d_in[b+1], (const float*)d_in[b+2],
                (const float*)d_in[b+3], (const float*)d_in[b+4], (const float*)d_in[b+5],
                (const float*)d_in[b+6] };
}

extern "C" void kernel_launch(void* const* d_in, const int* in_sizes, int n_in,
                              void* d_out, int out_size, void* d_ws, size_t ws_size,
                              hipStream_t stream)
{
  (void)in_sizes; (void)n_in; (void)out_size; (void)ws_size;

  const float* node_feats = (const float*)d_in[0];   // [N0,128]
  const float* edge_feats = (const float*)d_in[1];   // [E0,128]
  const float* node_path0 = (const float*)d_in[2];   // [N1,128] (ini_y3)
  const float* edge_path0 = (const float*)d_in[3];   // [E1,64]
  EgatP L1  = getP(d_in, 4);
  EgatP L2a = getP(d_in, 11);
  EgatP L2b = getP(d_in, 18);
  EgatP L3  = getP(d_in, 25);
  const float* lin1_W = (const float*)d_in[32];      // [256,128]
  const float* lin1_b = (const float*)d_in[33];      // [128]
  const int* src0 = (const int*)d_in[34];
  const int* dst0 = (const int*)d_in[35];
  const int* src1 = (const int*)d_in[36];
  const int* dst1 = (const int*)d_in[37];

  float* ws  = (float*)d_ws;
  float* out = (float*)d_out;
  float* out_nf = out;                                           // [N0,128]
  float* out_ef = out + (size_t)GN0 * 128;                       // [E0,128]
  float* out_np = out + (size_t)GN0 * 128 + (size_t)GE0 * 128;   // [N1,128]
  float* out_ep = out_np + (size_t)GN1 * 128;                    // [E1,64]

  // ---- stage 1: EGAT on atom graph ----
  run_egat(node_feats, edge_feats, 1, src0, dst0, GN0, GE0, 128, 128, 128, 128,
           L1, ws, ws + oNF, ws + oEFS, stream);

  // ---- lin1: x11 == ef viewed [N1,256]; node_path = x11@W + b + ini_y3 ----
  gemm_wmma_kernel<<<dim3((GN1 + 63) / 64, 128 / 32), dim3(256), 0, stream>>>(
      ws + oEFS, 256, 1, lin1_W, lin1_b, node_path0, out_np, GN1, 256, 128);

  // ---- stage 2: two EGAT layers on path graph (node/edge live in d_out slices) ----
  run_egat(out_np, edge_path0, 1, src1, dst1, GN1, GE1, 128, 64, 128, 64,
           L2a, ws, out_np, out_ep, stream);
  run_egat(out_np, out_ep, 1, src1, dst1, GN1, GE1, 128, 64, 128, 64,
           L2b, ws, out_np, out_ep, stream);

  // ---- stage 3: EGAT on atom graph; efeats = repeat(node_path,2) via row/2 ----
  run_egat(ws + oNF, out_np, 2, src0, dst0, GN0, GE0, 128, 128, 128, 128,
           L3, ws, out_nf, out_ef, stream);
}
